// CrossAttentionFusion_38001870635066
// MI455X (gfx1250) — compile-verified, matrix operand order checked by reading, not run on hardware
//
#include <hip/hip_runtime.h>
#include <hip/hip_bf16.h>

typedef __bf16 bf16_t;
typedef bf16_t v16bf __attribute__((ext_vector_type(16)));
typedef bf16_t v8bf  __attribute__((ext_vector_type(8)));
typedef float  v8f   __attribute__((ext_vector_type(8)));

// Problem constants (from reference)
constexpr int Bb  = 4;
constexpr int S   = 2048;
constexpr int SEM = 768;
constexpr int GRA = 512;
constexpr int D   = 1024;    // OUT_DIM
constexpr int M   = Bb * S;  // 8192 total rows
constexpr int Tt  = S;       // 2048

// ---------------------------------------------------------------------------
// CDNA5 async global->LDS copy (ASYNCcnt-tracked) + waits, via inline asm
// ---------------------------------------------------------------------------
__device__ __forceinline__ void async_copy_b128(unsigned lds_byte_off, const void* gsrc)
{
    asm volatile("global_load_async_to_lds_b128 %0, %1, off"
                 :: "v"(lds_byte_off), "v"((unsigned long long)(uintptr_t)gsrc)
                 : "memory");
}
__device__ __forceinline__ void wait_async_all() {
    asm volatile("s_wait_asynccnt 0x0" ::: "memory");
}
__device__ __forceinline__ void wait_async_le4() {
    asm volatile("s_wait_asynccnt 0x4" ::: "memory");
}

// ---------------------------------------------------------------------------
// Generic NT GEMM:  C[M,N] = alpha * A[M,K] @ Bt[N,K]^T   (bf16 in, f32 out)
// 256 threads = 8 waves; block tile 128x128; wave tile 32x64 (2x4 WMMA tiles).
// K-step 32, double-buffered LDS staging via async global->LDS (16KB/step,
// zero redundancy). LDS rows padded to 40 elements (80B) -> conflict-free
// ds_load_b128 fragment reads. M,N multiples of 128; K multiple of 32.
// Dynamic LDS: 2 bufs * (A 128x40 + B 128x40) * 2B = 40960 bytes.
// ---------------------------------------------------------------------------
__global__ __launch_bounds__(256)
void gemm_nt_bf16(const bf16_t* __restrict__ A, const bf16_t* __restrict__ Bt,
                  float* __restrict__ C,
                  int Mi, int Ni, int Ki, float alpha,
                  long long strideA, long long strideB, long long strideC)
{
    extern __shared__ bf16_t smem[];  // [buf][A:128*40 | B:128*40]

    const int batch = blockIdx.z;
    A  += (long long)batch * strideA;
    Bt += (long long)batch * strideB;
    C  += (long long)batch * strideC;

    const int tid  = threadIdx.x;
    const int lane = tid & 31;
    const int wave = tid >> 5;   // 0..7
    const int wm   = wave >> 1;  // 0..3 (M dir)
    const int wn   = wave & 1;   // 0..1 (N dir)

    const int mBlock = blockIdx.y * 128;
    const int nBlock = blockIdx.x * 128;

    const int rsel = lane & 15;         // row (A) / col (B) within 16
    const int kh   = (lane >> 4) << 3;  // k-half element offset: 0 or 8

    // stage one 128x32 tile of A and of B into LDS buffer `buf` (async).
    // 512 16B chunks per operand; each thread issues 2 per operand (4 total).
    auto stage = [&](int buf, int k0) {
        const unsigned baseA = (unsigned)buf * 20480u;
        const unsigned baseB = baseA + 10240u;
        #pragma unroll
        for (int c = 0; c < 2; ++c) {
            const int chunk = tid * 2 + c;
            const int row   = chunk >> 2;   // 0..127
            const int q     = chunk & 3;    // 16B sub-chunk of the 64B row
            async_copy_b128(baseA + (unsigned)row * 80u + (unsigned)q * 16u,
                            A + (long long)(mBlock + row) * Ki + k0 + q * 8);
            async_copy_b128(baseB + (unsigned)row * 80u + (unsigned)q * 16u,
                            Bt + (long long)(nBlock + row) * Ki + k0 + q * 8);
        }
    };

    v8f acc[2][4];
    #pragma unroll
    for (int i = 0; i < 2; ++i)
        #pragma unroll
        for (int j = 0; j < 4; ++j)
            #pragma unroll
            for (int e = 0; e < 8; ++e) acc[i][j][e] = 0.0f;

    stage(0, 0);
    int buf = 0;
    for (int k0 = 0; k0 < Ki; k0 += 32) {
        if (k0 + 32 < Ki) { stage(buf ^ 1, k0 + 32); wait_async_le4(); }
        else              { wait_async_all(); }
        __syncthreads();   // all waves' stage for `buf` complete

        const bf16_t* sa = smem + buf * 10240 + (wm * 32) * 40;
        const bf16_t* sb = smem + buf * 10240 + 5120 + (wn * 64) * 40;

        union Frag { v16bf v; v8bf h[2]; };
        Frag af[2], bfr[4];
        #pragma unroll
        for (int i = 0; i < 2; ++i) {
            const bf16_t* p = sa + (i * 16 + rsel) * 40 + kh;
            af[i].h[0] = *(const v8bf*)(p);
            af[i].h[1] = *(const v8bf*)(p + 16);
        }
        #pragma unroll
        for (int j = 0; j < 4; ++j) {
            const bf16_t* p = sb + (j * 16 + rsel) * 40 + kh;
            bfr[j].h[0] = *(const v8bf*)(p);
            bfr[j].h[1] = *(const v8bf*)(p + 16);
        }
        #pragma unroll
        for (int i = 0; i < 2; ++i)
            #pragma unroll
            for (int j = 0; j < 4; ++j)
                acc[i][j] = __builtin_amdgcn_wmma_f32_16x16x32_bf16(
                    false, af[i].v, false, bfr[j].v, (short)0, acc[i][j], false, false);

        __syncthreads();   // all waves done reading `buf` before it is rewritten
        buf ^= 1;
    }

    // C/D layout: VGPR r -> M = r (lanes 0-15) or 8+r (lanes 16-31); N = lane&15
    const int moff = (lane >> 4) * 8;
    const int col  = lane & 15;
    #pragma unroll
    for (int i = 0; i < 2; ++i)
        #pragma unroll
        for (int j = 0; j < 4; ++j) {
            float* cp = C + (long long)(mBlock + wm * 32 + i * 16 + moff) * Ni
                          + (nBlock + wn * 64 + j * 16 + col);
            #pragma unroll
            for (int r = 0; r < 8; ++r)
                cp[(long long)r * Ni] = alpha * acc[i][j][r];
        }
    (void)Mi;
}

// ---------------------------------------------------------------------------
// Elementwise f32 -> bf16
// ---------------------------------------------------------------------------
__global__ void cvt_f32_bf16(const float* __restrict__ in, bf16_t* __restrict__ out, int n)
{
    int i = blockIdx.x * blockDim.x + threadIdx.x;
    if (i < n) out[i] = (bf16_t)in[i];
}

// W [K x N] f32 -> WT [N x K] bf16, LDS-tiled so both sides are coalesced.
// Grid: (N/32, K/32), block (32,8).
__global__ __launch_bounds__(256)
void transpose_cvt_w(const float* __restrict__ in, bf16_t* __restrict__ out,
                     int Kd, int Nd)
{
    __shared__ bf16_t tile[32][33];
    const int n0 = blockIdx.x * 32, k0 = blockIdx.y * 32;
    for (int i = threadIdx.y; i < 32; i += 8)
        tile[i][threadIdx.x] = (bf16_t)in[(long long)(k0 + i) * Nd + n0 + threadIdx.x];
    __syncthreads();
    for (int i = threadIdx.y; i < 32; i += 8)
        out[(long long)(n0 + i) * Kd + k0 + threadIdx.x] = tile[threadIdx.x][i];
}

// Batched bf16 transpose: in [R x Cc] -> out [Cc x R] per batch (R,Cc mult of 32)
__global__ __launch_bounds__(256)
void transpose_bf16(const bf16_t* __restrict__ in, bf16_t* __restrict__ out, int R, int Cc)
{
    __shared__ bf16_t tile[32][33];
    const long long bo = (long long)blockIdx.z * R * Cc;
    in += bo; out += bo;
    const int c0 = blockIdx.x * 32, r0 = blockIdx.y * 32;
    for (int i = threadIdx.y; i < 32; i += 8)
        tile[i][threadIdx.x] = in[(long long)(r0 + i) * Cc + c0 + threadIdx.x];
    __syncthreads();
    for (int i = threadIdx.y; i < 32; i += 8)
        out[(long long)(c0 + i) * R + r0 + threadIdx.x] = tile[threadIdx.x][i];
}

// ---------------------------------------------------------------------------
// Bias + LayerNorm per row; writes f32 and bf16 copies
// ---------------------------------------------------------------------------
__global__ __launch_bounds__(256)
void bias_layernorm(const float* __restrict__ X, const float* __restrict__ bias,
                    const float* __restrict__ gamma, const float* __restrict__ beta,
                    float* __restrict__ Yf, bf16_t* __restrict__ Yb, int ncols)
{
    const long long row = blockIdx.x;
    const float* x = X + row * ncols;
    __shared__ float red[256];
    const int tid = threadIdx.x;

    float lsum = 0.f, lsq = 0.f;
    for (int c = tid; c < ncols; c += 256) {
        float v = x[c] + bias[c];
        lsum += v; lsq += v * v;
    }
    red[tid] = lsum; __syncthreads();
    for (int s = 128; s > 0; s >>= 1) { if (tid < s) red[tid] += red[tid + s]; __syncthreads(); }
    const float mean = red[0] / (float)ncols;
    __syncthreads();
    red[tid] = lsq; __syncthreads();
    for (int s = 128; s > 0; s >>= 1) { if (tid < s) red[tid] += red[tid + s]; __syncthreads(); }
    const float var  = red[0] / (float)ncols - mean * mean;
    const float rstd = rsqrtf(var + 1e-5f);

    for (int c = tid; c < ncols; c += 256) {
        float v = (x[c] + bias[c] - mean) * rstd * gamma[c] + beta[c];
        Yf[row * ncols + c] = v;
        Yb[row * ncols + c] = (bf16_t)v;
    }
}

// ---------------------------------------------------------------------------
// Row softmax (over last dim) -> bf16
// ---------------------------------------------------------------------------
__global__ __launch_bounds__(256)
void softmax_row(const float* __restrict__ X, bf16_t* __restrict__ Y, int ncols)
{
    const long long row = blockIdx.x;
    const float* x = X + row * ncols;
    __shared__ float red[256];
    const int tid = threadIdx.x;

    float lmax = -3.4e38f;
    for (int c = tid; c < ncols; c += 256) lmax = fmaxf(lmax, x[c]);
    red[tid] = lmax; __syncthreads();
    for (int s = 128; s > 0; s >>= 1) { if (tid < s) red[tid] = fmaxf(red[tid], red[tid + s]); __syncthreads(); }
    const float mx = red[0];
    __syncthreads();

    float lsum = 0.f;
    for (int c = tid; c < ncols; c += 256) lsum += __expf(x[c] - mx);
    red[tid] = lsum; __syncthreads();
    for (int s = 128; s > 0; s >>= 1) { if (tid < s) red[tid] += red[tid + s]; __syncthreads(); }
    const float inv = 1.0f / red[0];

    for (int c = tid; c < ncols; c += 256)
        Y[row * ncols + c] = (bf16_t)(__expf(x[c] - mx) * inv);
}

// Column softmax (over dim -2), written TRANSPOSED: Y[t,s] = softmax_s(X[s,t]).
// Block (32,8) owns 32 columns; all global accesses coalesced; 32x32 LDS
// transpose for the output pass.
__global__ __launch_bounds__(256)
void softmax_col_T(const float* __restrict__ X, bf16_t* __restrict__ Y, int nrows, int ncols)
{
    const long long bo = (long long)blockIdx.y * nrows * ncols;
    const float* x = X + bo;
    bf16_t* y = Y + bo;
    const int t0 = blockIdx.x * 32;
    const int tx = threadIdx.x, ty = threadIdx.y;

    __shared__ float red[8][32];
    __shared__ float cmax[32], cinv[32];
    __shared__ float tile[32][33];

    float lmax = -3.4e38f;
    for (int s = ty; s < nrows; s += 8)
        lmax = fmaxf(lmax, x[(long long)s * ncols + t0 + tx]);
    red[ty][tx] = lmax; __syncthreads();
    if (ty == 0) {
        float m = red[0][tx];
        #pragma unroll
        for (int r = 1; r < 8; ++r) m = fmaxf(m, red[r][tx]);
        cmax[tx] = m;
    }
    __syncthreads();
    const float mx = cmax[tx];

    float lsum = 0.f;
    for (int s = ty; s < nrows; s += 8)
        lsum += __expf(x[(long long)s * ncols + t0 + tx] - mx);
    red[ty][tx] = lsum; __syncthreads();
    if (ty == 0) {
        float sum = 0.f;
        #pragma unroll
        for (int r = 0; r < 8; ++r) sum += red[r][tx];
        cinv[tx] = 1.0f / sum;
    }
    __syncthreads();

    for (int s0 = 0; s0 < nrows; s0 += 32) {
        #pragma unroll
        for (int r = 0; r < 4; ++r) {
            const int i = ty + r * 8;
            tile[i][tx] = x[(long long)(s0 + i) * ncols + t0 + tx];
        }
        __syncthreads();
        #pragma unroll
        for (int r = 0; r < 4; ++r) {
            const int i = ty + r * 8;
            y[(long long)(t0 + i) * nrows + s0 + tx] =
                (bf16_t)(__expf(tile[tx][i] - cmax[i]) * cinv[i]);
        }
        __syncthreads();
    }
}

// fused[m, 0:D] = bf16(sem + ctx1); fused[m, D:2D] = bf16(gra + ctx2)
__global__ void fuse_add(const float* __restrict__ sem, const float* __restrict__ ctx1,
                         const float* __restrict__ gra, const float* __restrict__ ctx2,
                         bf16_t* __restrict__ fused, int total)   // total = M*D
{
    int i = blockIdx.x * blockDim.x + threadIdx.x;
    if (i < total) {
        int m = i / D, d = i % D;
        fused[(long long)m * (2 * D) + d]     = (bf16_t)(sem[i] + ctx1[i]);
        fused[(long long)m * (2 * D) + D + d] = (bf16_t)(gra[i] + ctx2[i]);
    }
}

__global__ void bias_relu(const float* __restrict__ x, const float* __restrict__ b,
                          float* __restrict__ out, int total, int Nd)
{
    int i = blockIdx.x * blockDim.x + threadIdx.x;
    if (i < total) out[i] = fmaxf(x[i] + b[i % Nd], 0.0f);
}

// ---------------------------------------------------------------------------
extern "C" void kernel_launch(void* const* d_in, const int* in_sizes, int n_in,
                              void* d_out, int out_size, void* d_ws, size_t ws_size,
                              hipStream_t stream)
{
    const float* semF   = (const float*)d_in[0];
    const float* graF   = (const float*)d_in[1];
    const float* W_sp   = (const float*)d_in[2];
    const float* b_sp   = (const float*)d_in[3];
    const float* W_gp   = (const float*)d_in[4];
    const float* b_gp   = (const float*)d_in[5];
    const float* gamma1 = (const float*)d_in[6];
    const float* beta1  = (const float*)d_in[7];
    const float* gamma2 = (const float*)d_in[8];
    const float* beta2  = (const float*)d_in[9];
    const float* W_out  = (const float*)d_in[10];
    const float* b_out  = (const float*)d_in[11];
    float* out = (float*)d_out;

    char* ws = (char*)d_ws;
    size_t off = 0;
    auto take = [&](size_t bytes) { size_t o = off; off += (bytes + 255) & ~size_t(255); return o; };

    bf16_t* semin_bf = (bf16_t*)(ws + take((size_t)M * SEM * 2));
    bf16_t* grain_bf = (bf16_t*)(ws + take((size_t)M * GRA * 2));
    bf16_t* WspT     = (bf16_t*)(ws + take((size_t)D * SEM * 2));
    bf16_t* WgpT     = (bf16_t*)(ws + take((size_t)D * GRA * 2));
    bf16_t* WoutT    = (bf16_t*)(ws + take((size_t)D * 2 * D * 2));
    float*  lin1     = (float*) (ws + take((size_t)M * D * 4));   // reused as sem_ctx
    float*  lin2     = (float*) (ws + take((size_t)M * D * 4));   // reused as gra_ctx
    float*  sem_f32  = (float*) (ws + take((size_t)M * D * 4));
    float*  gra_f32  = (float*) (ws + take((size_t)M * D * 4));
    bf16_t* sem_bf   = (bf16_t*)(ws + take((size_t)M * D * 2));
    bf16_t* gra_bf   = (bf16_t*)(ws + take((size_t)M * D * 2));
    bf16_t* semT_bf  = (bf16_t*)(ws + take((size_t)M * D * 2));   // [b][D][S]
    bf16_t* graT_bf  = (bf16_t*)(ws + take((size_t)M * D * 2));   // [b][D][T]
    float*  attn     = (float*) (ws + take((size_t)Bb * S * Tt * 4)); // reused as final lin
    bf16_t* sms_bf   = (bf16_t*)(ws + take((size_t)Bb * S * Tt * 2));
    bf16_t* smgT_bf  = (bf16_t*)(ws + take((size_t)Bb * S * Tt * 2)); // [b][T][S]
    bf16_t* fused_bf = (bf16_t*)(ws + take((size_t)M * 2 * D * 2));

    const int thr = 256;
    const int GEMM_LDS = 40960;
    auto nb = [&](int n) { return (n + thr - 1) / thr; };

    // 1) bf16 conversions of activations + transposed bf16 weights
    cvt_f32_bf16<<<nb(M * SEM), thr, 0, stream>>>(semF, semin_bf, M * SEM);
    cvt_f32_bf16<<<nb(M * GRA), thr, 0, stream>>>(graF, grain_bf, M * GRA);
    transpose_cvt_w<<<dim3(D / 32, SEM / 32), dim3(32, 8), 0, stream>>>(W_sp, WspT, SEM, D);
    transpose_cvt_w<<<dim3(D / 32, GRA / 32), dim3(32, 8), 0, stream>>>(W_gp, WgpT, GRA, D);
    transpose_cvt_w<<<dim3(D / 32, (2 * D) / 32), dim3(32, 8), 0, stream>>>(W_out, WoutT, 2 * D, D);

    // 2) Projections
    gemm_nt_bf16<<<dim3(D / 128, M / 128, 1), 256, GEMM_LDS, stream>>>(
        semin_bf, WspT, lin1, M, D, SEM, 1.0f, 0, 0, 0);
    gemm_nt_bf16<<<dim3(D / 128, M / 128, 1), 256, GEMM_LDS, stream>>>(
        grain_bf, WgpT, lin2, M, D, GRA, 1.0f, 0, 0, 0);

    // 3) Bias + LayerNorm (f32 + bf16 copies)
    bias_layernorm<<<M, 256, 0, stream>>>(lin1, b_sp, gamma1, beta1, sem_f32, sem_bf, D);
    bias_layernorm<<<M, 256, 0, stream>>>(lin2, b_gp, gamma2, beta2, gra_f32, gra_bf, D);

    // 4) Transposed bf16 copies for ctx GEMMs
    transpose_bf16<<<dim3(D / 32, S / 32, Bb), dim3(32, 8), 0, stream>>>(sem_bf, semT_bf, S, D);
    transpose_bf16<<<dim3(D / 32, S / 32, Bb), dim3(32, 8), 0, stream>>>(gra_bf, graT_bf, S, D);

    // 5) attn = sem @ gra^T / 32   (NT with Bt = gra rows directly)
    gemm_nt_bf16<<<dim3(Tt / 128, S / 128, Bb), 256, GEMM_LDS, stream>>>(
        sem_bf, gra_bf, attn, S, Tt, D, 0.03125f,
        (long long)S * D, (long long)Tt * D, (long long)S * Tt);

    // 6) Softmaxes
    softmax_row<<<Bb * S, 256, 0, stream>>>(attn, sms_bf, Tt);
    softmax_col_T<<<dim3(Tt / 32, Bb), dim3(32, 8), 0, stream>>>(attn, smgT_bf, S, Tt);

    // 7) sem_ctx = sm_s @ gra  (Bt = graT);  gra_ctx = sm_g^T @ sem (Bt = semT)
    gemm_nt_bf16<<<dim3(D / 128, S / 128, Bb), 256, GEMM_LDS, stream>>>(
        sms_bf, graT_bf, lin1, S, D, Tt, 1.0f,
        (long long)S * Tt, (long long)D * Tt, (long long)S * D);
    gemm_nt_bf16<<<dim3(D / 128, Tt / 128, Bb), 256, GEMM_LDS, stream>>>(
        smgT_bf, semT_bf, lin2, Tt, D, S, 1.0f,
        (long long)Tt * S, (long long)D * S, (long long)Tt * D);

    // 8) fused = concat(sem + sem_ctx, gra + gra_ctx) -> bf16
    fuse_add<<<nb(M * D), thr, 0, stream>>>(sem_f32, lin1, gra_f32, lin2, fused_bf, M * D);

    // 9) out = relu(fused @ W_out + b_out)  (reuse attn buffer for pre-relu f32)
    gemm_nt_bf16<<<dim3(D / 128, M / 128, 1), 256, GEMM_LDS, stream>>>(
        fused_bf, WoutT, attn, M, D, 2 * D, 1.0f, 0, 0, 0);
    bias_relu<<<nb(M * D), thr, 0, stream>>>(attn, b_out, out, M * D, D);

    (void)in_sizes; (void)n_in; (void)out_size; (void)ws_size;
}
